// Dereverb_T60_module_39805756899483
// MI455X (gfx1250) — matrix operand order repaired and checked by reading, not compiled, not run on hardware
//
#include <hip/hip_runtime.h>

// CDNA5 / gfx1250, wave32. Batched bidirectional GRU (H=16, input dim 1) over
// 8000 independent windows; 16 sequences per wave, gates computed with
// v_wmma_f32_16x16x32_f16 (A = augmented gate weights, B = [h | x | 1 | 0]).

typedef _Float16 v16h __attribute__((ext_vector_type(16)));
typedef float    v8f  __attribute__((ext_vector_type(8)));
typedef _Float16 h2   __attribute__((ext_vector_type(2)));

__device__ __forceinline__ float sigmoid_f(float x) {
  return __builtin_amdgcn_rcpf(1.0f + __expf(-x));   // v_exp_f32 + v_rcp_f32
}
__device__ __forceinline__ float tanh_f(float x) {
  return __builtin_fmaf(2.0f, __builtin_amdgcn_rcpf(1.0f + __expf(-2.0f * x)), -1.0f);
}
__device__ __forceinline__ unsigned pk16(float lo, float hi) {
  h2 p; p.x = (_Float16)lo; p.y = (_Float16)hi;
  return __builtin_bit_cast(unsigned, p);
}
// exchange lanes [0..15] <-> [16..31]: group-of-32 swizzle, xor=0x10, and=0x1f
__device__ __forceinline__ unsigned swz_hi16(unsigned x) {
  return (unsigned)__builtin_amdgcn_ds_swizzle((int)x, 0x401F);
}
__device__ __forceinline__ float swz_hi16f(float x) {
  return __builtin_bit_cast(float, swz_hi16(__builtin_bit_cast(unsigned, x)));
}

// A-matrix (16x32 f16) layout: lanes 0-15 / 16-31 both hold M=0..15;
// element e of v16h -> K = e + 16*laneHalf (contiguous k order).
// Gate rows: K<16 = W_hh[row][K]; K==16 = w_ih[row] (0 for n-gate);
// K==17 = bias (b_ih+b_hh for r/z, b_hh for n); K>=18 = 0.
__device__ __forceinline__ void load_weights(
    const float* __restrict__ w_ih, const float* __restrict__ w_hh,
    const float* __restrict__ b_ih, const float* __restrict__ b_hh,
    int m, int gsel, v16h& Ar, v16h& Az, v16h& An, float wn[8], float bn[8])
{
  union U { v16h v; _Float16 h[16]; } A[3];
  #pragma unroll
  for (int G = 0; G < 3; ++G) {
    const bool ng = (G == 2);
    const int row = G * 16 + m;
    #pragma unroll
    for (int e = 0; e < 16; ++e) {
      const int K = e + 16 * gsel;
      float val;
      if (K < 16)       val = w_hh[row * 16 + K];
      else if (K == 16) val = ng ? 0.0f : w_ih[row];
      else if (K == 17) val = ng ? b_hh[row] : (b_ih[row] + b_hh[row]);
      else              val = 0.0f;
      A[G].h[e] = (_Float16)val;
    }
  }
  Ar = A[0].v; Az = A[1].v; An = A[2].v;
  // per-lane constants for i_n (D layout: VGPR v -> hidden v + 8*laneHalf)
  #pragma unroll
  for (int v = 0; v < 8; ++v) {
    const int row = 32 + v + 8 * gsel;
    wn[v] = w_ih[row];
    bn[v] = b_ih[row];
  }
}

// One GRU step for 16 sequences. hD[8]: f32 state in WMMA C/D layout
// (lane = seq%16, VGPR v + 8*laneHalf = hidden unit).
__device__ __forceinline__ void gru_step(
    float hD[8], float x, bool mm, bool lo,
    v16h Ar, v16h Az, v16h An, const float wn[8], const float bn[8])
{
  // Repack state into B layout (32x16 f16): lanes 0-15 K=0..15 (K=2v+half),
  // lanes 16-31 K=16..31 (row16 = x, row17 = 1, rest 0).
  unsigned P[4];
  #pragma unroll
  for (int j = 0; j < 4; ++j) P[j] = pk16(hD[2 * j], hD[2 * j + 1]);
  unsigned Q[4];
  #pragma unroll
  for (int j = 0; j < 4; ++j) Q[j] = swz_hi16(P[j]);   // hidden 8..15 -> lanes 0-15
  const unsigned xpair = pk16(x, 1.0f);
  union { v16h v; unsigned u[8]; } B;
  B.u[0] = lo ? P[0] : xpair;
  #pragma unroll
  for (int j = 1; j < 4; ++j) B.u[j] = lo ? P[j] : 0u;
  #pragma unroll
  for (int j = 0; j < 4; ++j) B.u[4 + j] = lo ? Q[j] : 0u;

  v8f zero = {0.f, 0.f, 0.f, 0.f, 0.f, 0.f, 0.f, 0.f};
  v8f Dr = __builtin_amdgcn_wmma_f32_16x16x32_f16(false, Ar, false, B.v, (short)0, zero, false, false);
  v8f Dz = __builtin_amdgcn_wmma_f32_16x16x32_f16(false, Az, false, B.v, (short)0, zero, false, false);
  v8f Dn = __builtin_amdgcn_wmma_f32_16x16x32_f16(false, An, false, B.v, (short)0, zero, false, false);

  #pragma unroll
  for (int v = 0; v < 8; ++v) {
    const float r  = sigmoid_f(Dr[v]);
    const float z  = sigmoid_f(Dz[v]);
    const float nn = tanh_f(__builtin_fmaf(x, wn[v], bn[v]) + r * Dn[v]);
    const float hn = nn + z * (hD[v] - nn);     // (1-z)*n + z*h
    hD[v] = mm ? hn : hD[v];                    // pad mask: keep h
  }
}

// mean numerator: sum of the 16 hidden units of this lane's sequence
__device__ __forceinline__ float hsum16(const float hD[8]) {
  float s = ((hD[0] + hD[1]) + (hD[2] + hD[3])) + ((hD[4] + hD[5]) + (hD[6] + hD[7]));
  return s + swz_hi16f(s);
}

__global__ __launch_bounds__(32) void dereverb_bigru_kernel(
    const float* __restrict__ input, const float* __restrict__ t60s,
    const float* __restrict__ w_ih_f, const float* __restrict__ w_hh_f,
    const float* __restrict__ b_ih_f, const float* __restrict__ b_hh_f,
    const float* __restrict__ w_ih_b, const float* __restrict__ w_hh_b,
    const float* __restrict__ b_ih_b, const float* __restrict__ b_hh_b,
    float* __restrict__ out, int T, int C, int tilesPerB)
{
  const int lane = threadIdx.x & 31;
  const int m    = lane & 15;        // sequence-in-tile / A-matrix row
  const int gsel = lane >> 4;        // lane half
  const bool lo  = (gsel == 0);

  const int b = blockIdx.x / tilesPerB;
  const int c = (blockIdx.x % tilesPerB) * 16 + m;   // chunk index
  const bool valid = (c < C);
  const int ce = valid ? c : (C - 1);

  const int t60 = (int)(t60s[0] * 1000.0f + 0.5f);
  const int S1  = t60 - 16;                          // warmup length
  int pad = ce * 16 + t60 - T; if (pad < 0) pad = 0; // pad <= S1 always
  const float* inb = input + (size_t)b * T;
  const int xbase = T - 1 - ce * 16 + pad;           // x(t) = inb[xbase - t], t >= pad

  v16h Ar, Az, An;
  float wn[8], bn[8];
  load_weights(w_ih_f, w_hh_f, b_ih_f, b_hh_f, m, gsel, Ar, Az, An, wn, bn);

  float hD[8];
  #pragma unroll
  for (int v = 0; v < 8; ++v) hD[v] = 0.0f;

  // ---------- forward warmup: t in [0, S1), x prefetched one step ahead ----------
  bool  mn = (0 >= pad);
  float xn = inb[xbase - (mn ? 0 : pad)];
  for (int t = 0; t < S1; ++t) {
    const float x  = mn ? xn : 0.0f;
    const bool  mm = mn;
    const int t1 = t + 1;
    mn = (t1 >= pad);
    xn = inb[xbase - (mn ? t1 : pad)];               // always in-bounds
    gru_step(hD, x, mm, lo, Ar, Az, An, wn, bn);
  }

  // ---------- forward collect: t = S1 + s (t >= pad guaranteed) ----------
  float outv[16];
  #pragma unroll
  for (int s = 0; s < 16; ++s) {
    const int t = S1 + s;
    const float x = inb[xbase - t];
    gru_step(hD, x, true, lo, Ar, Az, An, wn, bn);
    outv[s] = hsum16(hD);
  }

  // ---------- backward: 16 steps, t = t60-1 down to S1 ----------
  load_weights(w_ih_b, w_hh_b, b_ih_b, b_hh_b, m, gsel, Ar, Az, An, wn, bn);
  #pragma unroll
  for (int v = 0; v < 8; ++v) hD[v] = 0.0f;
  #pragma unroll
  for (int s = 15; s >= 0; --s) {
    const int t = S1 + s;
    const float x = inb[xbase - t];
    gru_step(hD, x, true, lo, Ar, Az, An, wn, bn);
    outv[s] += hsum16(hD);                           // ys_b reversed alignment
  }

  // ---------- store: final[b, T-1-(c*16+s)] = (sumF+sumB)/16 ----------
  if (lo && valid) {
    float* ob = out + (size_t)b * T + (T - 16 - c * 16);  // 64B aligned
    #pragma unroll
    for (int k = 0; k < 4; ++k) {
      float4 stv = make_float4(outv[15 - 4 * k] * 0.0625f,
                               outv[14 - 4 * k] * 0.0625f,
                               outv[13 - 4 * k] * 0.0625f,
                               outv[12 - 4 * k] * 0.0625f);
      *(float4*)(ob + 4 * k) = stv;
    }
  }
}

extern "C" void kernel_launch(void* const* d_in, const int* in_sizes, int n_in,
                              void* d_out, int out_size, void* d_ws, size_t ws_size,
                              hipStream_t stream) {
  const int B = 8;                       // fixed by setup_inputs
  const int T = in_sizes[0] / B;         // 16000
  const int C = (T + 15) / 16;           // chunks per batch row
  const int tilesPerB = (C + 15) / 16;   // 16 sequences per wave
  dim3 grid(B * tilesPerB), block(32);   // one wave per block; EXEC all-ones
  dereverb_bigru_kernel<<<grid, block, 0, stream>>>(
      (const float*)d_in[0], (const float*)d_in[1],
      (const float*)d_in[2], (const float*)d_in[3],
      (const float*)d_in[4], (const float*)d_in[5],
      (const float*)d_in[6], (const float*)d_in[7],
      (const float*)d_in[8], (const float*)d_in[9],
      (float*)d_out, T, C, tilesPerB);
}
